// BayerNN_2302102471018
// MI455X (gfx1250) — matrix operand order repaired and checked by reading, not compiled, not run on hardware
//
#include <hip/hip_runtime.h>
#include <hip/hip_bf16.h>

typedef _Float16 v8h  __attribute__((ext_vector_type(8)));
typedef _Float16 v16h __attribute__((ext_vector_type(16)));
typedef float    v8f  __attribute__((ext_vector_type(8)));

#define H2  252
#define FOV 5

// ---------------- weight prep: f32 row-major (K x N) -> f16 col-major padded (Np x Kp)
__global__ void prep_weights_kernel(const float* __restrict__ W, _Float16* __restrict__ out,
                                    int Kreal, int Nreal, int Kp, int Np) {
  int idx = blockIdx.x * blockDim.x + threadIdx.x;
  if (idx >= Kp * Np) return;
  int n = idx / Kp, k = idx - n * Kp;
  float v = (k < Kreal && n < Nreal) ? W[k * Nreal + n] : 0.0f;
  out[n * Kp + k] = (_Float16)v;
}

__global__ void prep_b5_kernel(const float* __restrict__ b5, float* __restrict__ b5p) {
  int i = threadIdx.x;
  if (i < 16) b5p[i] = (i < 8) ? b5[i] : 0.0f;
}

// ---------------- one MLP layer: act(16 x KT*32, f16 in LDS) @ W(KT*32 x NT*16) + b, lrelu,
// stored back in-place into the wave-private LDS region with the new stride.
template<int KT, int NT>
__device__ __forceinline__ void layer_fwd(_Float16* act, int sIn, int sOut,
                                          const _Float16* __restrict__ wgt,
                                          const float* __restrict__ bias,
                                          int lane) {
  const int mA = lane & 15;     // A row (both half-waves carry rows 0..15)
  const int hs = lane >> 4;     // which K-half of the lane group
  const int nc = lane & 15;     // B/C column within tile
  constexpr int Kp = KT * 32;

  v8f acc[NT];
#pragma unroll
  for (int n = 0; n < NT; ++n) {
    float bv = bias[n * 16 + nc];
    v8f c;
#pragma unroll
    for (int i = 0; i < 8; ++i) c[i] = bv;
    acc[n] = c;
  }

#pragma unroll
  for (int k = 0; k < KT; ++k) {
    // A layout (matches B): lanes<16 hold K 0..15, lanes>=16 hold K 16..31
    const int abase = mA * sIn + k * 32 + hs * 16;
    v8h a0 = *(const v8h*)(act + abase);
    v8h a1 = *(const v8h*)(act + abase + 8);
    v16h A;
#pragma unroll
    for (int i = 0; i < 8; ++i) { A[i] = a0[i]; A[i + 8] = a1[i]; }
#pragma unroll
    for (int n = 0; n < NT; ++n) {
      // B layout (matches A): col = lane&15, contiguous K halves hs*16..hs*16+15
      v16h B = *(const v16h*)(wgt + (size_t)(n * 16 + nc) * Kp + k * 32 + hs * 16);
      acc[n] = __builtin_amdgcn_wmma_f32_16x16x32_f16(false, A, false, B,
                                                      (short)0, acc[n], false, false);
    }
  }

  // leaky-relu + f16 store (C layout: VGPR r -> row r + hs*8, col = lane&15 + 16*n)
#pragma unroll
  for (int n = 0; n < NT; ++n) {
#pragma unroll
    for (int r = 0; r < 8; ++r) {
      float v = acc[n][r];
      v = (v >= 0.0f) ? v : 0.01f * v;
      act[(r + hs * 8) * sOut + n * 16 + nc] = (_Float16)v;
    }
  }
}

// ---------------- fused demosaic + MLP kernel
__global__ __launch_bounds__(256) void bayer_mlp_kernel(
    const float* __restrict__ mosaic,
    const _Float16* __restrict__ w1, const _Float16* __restrict__ w2,
    const _Float16* __restrict__ w3, const _Float16* __restrict__ w4,
    const _Float16* __restrict__ w5,
    const float* __restrict__ b1, const float* __restrict__ b2,
    const float* __restrict__ b3, const float* __restrict__ b4,
    const float* __restrict__ b5p,
    float* __restrict__ out) {

  __shared__ __align__(32) _Float16 s_act[128 * 128];   // 32 KB: 128 pixels x 128 feat (f16)
  __shared__ float s_cswin[4 * 12 * 20];                // 4 phases, (8+4) x (16+4) window
  __shared__ float s_nrm[128];

  const int t    = threadIdx.x;
  const int lane = t & 31;
  const int wave = t >> 5;
  const int x0 = blockIdx.x * 16;
  const int y0 = blockIdx.y * 8;
  const int b  = blockIdx.z;
  const float* __restrict__ mb = mosaic + (size_t)b * 3 * 512 * 512;

  __builtin_prefetch(w1, 0, 1);   // global_prefetch_b8: warm L2 for first-layer weights

  // 1) build phase-sum window in LDS (g = sum over 3 channels), clamped at borders
  for (int e = t; e < 960; e += 256) {
    int c  = e / 240;
    int rr = e - c * 240;
    int wy = rr / 20;
    int wx = rr - wy * 20;
    int gy = 2 * (y0 + wy) + (c & 1);  if (gy > 511) gy = 511;
    int gx = 2 * (x0 + wx) + (c >> 1); if (gx > 511) gx = 511;
    int gi = gy * 512 + gx;
    s_cswin[e] = mb[gi] + mb[262144 + gi] + mb[524288 + gi];
  }
  __syncthreads();

  // 2) per-pixel 100-dim feature build + mean-normalize -> f16 rows in s_act
  if (t < 128) {
    const int ty = t >> 4, tx = t & 15;
    float ssum = 0.0f;
#pragma unroll
    for (int c = 0; c < 4; ++c)
#pragma unroll
      for (int i = 0; i < FOV; ++i)
#pragma unroll
        for (int j = 0; j < FOV; ++j)
          ssum += s_cswin[(c * 12 + ty + i) * 20 + tx + j];
    float nrm = ssum * 0.01f + 1e-8f;
    s_nrm[t] = nrm;
    float inv = 1.0f / nrm;
    _Float16* arow = s_act + t * 128;
    int f = 0;
#pragma unroll
    for (int c = 0; c < 4; ++c)
#pragma unroll
      for (int i = 0; i < FOV; ++i)
#pragma unroll
        for (int j = 0; j < FOV; ++j)
          arow[f++] = (_Float16)(s_cswin[(c * 12 + ty + i) * 20 + tx + j] * inv);
#pragma unroll
    for (int z = 100; z < 128; ++z) arow[z] = (_Float16)0.0f;
  }
  __syncthreads();

  // 3) MLP: each wave owns 16 pixels (one y row of the tile), wave-private LDS region
  _Float16* act = s_act + wave * (16 * 128);

  layer_fwd<4, 8>(act, 128, 128, w1, b1, lane);   // 100(pad128) -> 128
  layer_fwd<4, 4>(act, 128, 64,  w2, b2, lane);   // 128 -> 64
  layer_fwd<2, 4>(act, 64,  64,  w3, b3, lane);   // 64 -> 64
  layer_fwd<2, 2>(act, 64,  32,  w4, b4, lane);   // 64 -> 32

  // 4) layer 5 (32 -> 8, N padded to 16) kept in registers
  const int mA = lane & 15, hs = lane >> 4, nc = lane & 15;
  v8f c5;
  {
    float bv = b5p[nc];
#pragma unroll
    for (int i = 0; i < 8; ++i) c5[i] = bv;
  }
  {
    const int abase = mA * 32 + hs * 16;
    v8h a0 = *(const v8h*)(act + abase);
    v8h a1 = *(const v8h*)(act + abase + 8);
    v16h A;
#pragma unroll
    for (int i = 0; i < 8; ++i) { A[i] = a0[i]; A[i + 8] = a1[i]; }
    v16h B = *(const v16h*)(w5 + (size_t)nc * 32 + hs * 16);
    c5 = __builtin_amdgcn_wmma_f32_16x16x32_f16(false, A, false, B, (short)0, c5, false, false);
  }

  // 5) scatter: cols 0..7 carry MLP outputs (x nrm); cols 8..11 do mosaic copies
  const int yy = y0 + wave;
  if (yy < H2) {
    float* __restrict__ ob = out + (size_t)b * 3 * 504 * 504;
#pragma unroll
    for (int r = 0; r < 8; ++r) {
      const int m = r + hs * 8;       // pixel within wave (C-layout row)
      const int x = x0 + m;
      if (x < H2) {
        if (nc < 8) {
          const float v  = c5[r] * s_nrm[wave * 16 + m];
          const int ch = (int)((0x20102120u >> (nc * 4)) & 0xF); // 0,2,1,2,0,1,0,2
          const int oy = 2 * yy + (nc >> 2);
          const int ox = 2 * x  + ((nc >> 1) & 1);
          ob[((size_t)ch * 504 + oy) * 504 + ox] = v;
        } else if (nc < 12) {
          const int q  = nc - 8;
          const int ch = (int)((0x1201u >> (q * 4)) & 0xF);      // 1,0,2,1
          const int dy = q >> 1, dx = q & 1;
          const float v = mb[(size_t)ch * 262144 +
                             (size_t)(4 + 2 * yy + dy) * 512 + (4 + 2 * x + dx)];
          ob[((size_t)ch * 504 + (2 * yy + dy)) * 504 + (2 * x + dx)] = v;
        }
      }
    }
  }
}

extern "C" void kernel_launch(void* const* d_in, const int* in_sizes, int n_in,
                              void* d_out, int out_size, void* d_ws, size_t ws_size,
                              hipStream_t stream) {
  const float* mosaic = (const float*)d_in[0];
  const float* W1 = (const float*)d_in[1];  const float* b1 = (const float*)d_in[2];
  const float* W2 = (const float*)d_in[3];  const float* b2 = (const float*)d_in[4];
  const float* W3 = (const float*)d_in[5];  const float* b3 = (const float*)d_in[6];
  const float* W4 = (const float*)d_in[7];  const float* b4 = (const float*)d_in[8];
  const float* W5 = (const float*)d_in[9];  const float* b5 = (const float*)d_in[10];
  float* out = (float*)d_out;

  char* ws = (char*)d_ws;
  _Float16* w1h = (_Float16*)(ws + 0);       // 128x128 f16 = 32768 B
  _Float16* w2h = (_Float16*)(ws + 32768);   // 64x128       = 16384 B
  _Float16* w3h = (_Float16*)(ws + 49152);   // 64x64        =  8192 B
  _Float16* w4h = (_Float16*)(ws + 57344);   // 32x64        =  4096 B
  _Float16* w5h = (_Float16*)(ws + 61440);   // 16x32        =  1024 B
  float*    b5p = (float*)   (ws + 62464);   // 16 floats

  prep_weights_kernel<<<64, 256, 0, stream>>>(W1, w1h, 100, 128, 128, 128);
  prep_weights_kernel<<<32, 256, 0, stream>>>(W2, w2h, 128,  64, 128,  64);
  prep_weights_kernel<<<16, 256, 0, stream>>>(W3, w3h,  64,  64,  64,  64);
  prep_weights_kernel<<< 8, 256, 0, stream>>>(W4, w4h,  64,  32,  64,  32);
  prep_weights_kernel<<< 2, 256, 0, stream>>>(W5, w5h,  32,   8,  32,  16);
  prep_b5_kernel<<<1, 16, 0, stream>>>(b5, b5p);

  dim3 grid(16, 32, 8);  // ceil(252/16) x-tiles, ceil(252/8) y-tiles, 8 batches
  bayer_mlp_kernel<<<grid, 256, 0, stream>>>(mosaic, w1h, w2h, w3h, w4h, w5h,
                                             b1, b2, b3, b4, b5p, out);
}
